// GATLayer_14207751815179
// MI455X (gfx1250) — matrix operand order repaired and checked by reading, not compiled, not run on hardware
//
#include <hip/hip_runtime.h>
#include <hip/hip_bf16.h>

// ---------------- problem constants (match reference) ----------------
#define NN   50000
#define EE   800000
#define FIN  256
#define HH   8
#define FO   32
#define HFO  256          // H*FO
#define NCOL 512          // proj (256) + skip (256) fused GEMM output columns
#define XPAD 264          // LDS row stride in bf16 elems: 256 + 8 (16B TDM pad -> bank spread)

typedef __attribute__((ext_vector_type(16))) __bf16 v16bf;
typedef __attribute__((ext_vector_type(8)))  __bf16 v8bf;
typedef __attribute__((ext_vector_type(8)))  float  v8f;
typedef __attribute__((ext_vector_type(4)))  unsigned int u32x4;
typedef __attribute__((ext_vector_type(8)))  int i32x8;
typedef __attribute__((ext_vector_type(4)))  int i32x4;

// ---------------- helpers ----------------
__device__ __forceinline__ unsigned fmap(float f) {
    unsigned u = __float_as_uint(f);
    return u ^ ((u >> 31) ? 0xFFFFFFFFu : 0x80000000u);   // monotone map float->uint
}
__device__ __forceinline__ float funmap(unsigned u) {
    return __uint_as_float(u ^ ((u >> 31) ? 0x80000000u : 0xFFFFFFFFu));
}
__device__ __forceinline__ void atomic_fadd(float* p, float v) {
    __hip_atomic_fetch_add(p, v, __ATOMIC_RELAXED, __HIP_MEMORY_SCOPE_AGENT);
}

// ---------------- K0: f32 -> bf16 convert (grid-stride) ----------------
__global__ void cvt_bf16_kernel(const float* __restrict__ in, __bf16* __restrict__ out, long n) {
    long i = (long)blockIdx.x * blockDim.x + threadIdx.x;
    long stride = (long)gridDim.x * blockDim.x;
    for (; i < n; i += stride) out[i] = (__bf16)in[i];
}

// ---------------- K1: fused projection GEMM via WMMA bf16 + TDM ----------------
// C[N,512] = X[N,256](bf16) @ Wcat[512,256](bf16)^T, fp32 accum.
// cols 0..255 -> proj (ws), cols 256..511 -> skip (written straight into d_out).
// Block = 256 threads = 8 waves; block owns one 16-row M tile; wave w owns 64 cols.
// X tile (16x256 bf16 = 8KB) is DMA'd into LDS once per block by the Tensor Data
// Mover (with 16B row padding for bank spread); A fragments then come from LDS.
__global__ __launch_bounds__(256) void gemm_wmma_kernel(
        const __bf16* __restrict__ X, const __bf16* __restrict__ W,
        float* __restrict__ proj, float* __restrict__ skip, int nrows) {
    const int wave = threadIdx.x >> 5;
    const int lane = threadIdx.x & 31;
    const int mt   = blockIdx.x;                 // 16-row tile index
    const int hi   = lane >> 4;                  // half-wave select
    const int col0 = wave * 64;
    const int aoff = hi ? 16 : 0;                // contiguous k order used here (A and B): lanes>=16 hold K+16..K+31

    __shared__ __attribute__((aligned(16))) char shbuf[16 * XPAD * 2];  // 8448 B

#if __has_builtin(__builtin_amdgcn_tensor_load_to_lds)
    // ---- TDM: DMA the 16x256 bf16 X tile into LDS (issued by wave 0 only) ----
    if (threadIdx.x < 32) {
        const unsigned long long ga =
            (unsigned long long)(uintptr_t)X + (unsigned long long)mt * (16ull * FIN * 2ull);
        const unsigned ldsoff = (unsigned)(uintptr_t)shbuf;
        u32x4 g0;
        g0[0] = 1u;                                            // count=1 (valid user D#)
        g0[1] = ldsoff;                                        // lds_addr (bytes)
        g0[2] = (unsigned)(ga & 0xFFFFFFFFull);                // global_addr[31:0]
        g0[3] = (unsigned)((ga >> 32) & 0x1FFFFFFull)          // global_addr[56:32]
              | (2u << 30);                                    // type = 2 ("image")
        i32x8 g1;
        g1[0] = (1 << 16)            // data_size = 1 -> 2 bytes
              | (1 << 20)            // pad_enable
              | (6 << 22)            // pad_interval = 6 -> every 128 DWORDs (512B row)
              | (3 << 25);           // pad_amount  = 3 -> 4 DWORDs (16B)
        g1[1] = (int)((unsigned)FIN << 16);                    // tensor_dim0[15:0] = 256
        g1[2] = (int)(((unsigned)nrows & 0xFFFFu) << 16);      // tensor_dim1[15:0]
        g1[3] = (int)(((unsigned)nrows >> 16) & 0xFFFFu)       // tensor_dim1[31:16]
              | (int)(256u << 16);                             // tile_dim0 = 256
        g1[4] = 16;                                            // tile_dim1 = 16, tile_dim2 = 0
        g1[5] = FIN;                                           // tensor_dim0_stride = 256
        g1[6] = 0;
        g1[7] = 0;
        i32x4 z4 = {0, 0, 0, 0};
#if defined(__clang_major__) && (__clang_major__ >= 23)
        i32x8 z8 = {0, 0, 0, 0, 0, 0, 0, 0};
        __builtin_amdgcn_tensor_load_to_lds(g0, g1, z4, z4, z8, 0);
#else
        __builtin_amdgcn_tensor_load_to_lds(g0, g1, z4, z4, 0);
#endif
        __builtin_amdgcn_s_wait_tensorcnt(0);
    }
    __syncthreads();
    const __bf16* __restrict__ lx = (const __bf16*)shbuf;      // padded rows: XPAD elems
    const int lrow = (lane & 15) * XPAD;
#else
    // ---- fallback: direct global A loads (round-1 path) ----
    int rowA = (mt << 4) + (lane & 15);
    if (rowA >= nrows) rowA = nrows - 1;
    const __bf16* __restrict__ lx = X + (size_t)rowA * FIN;
    const int lrow = 0;
#endif

    v8f acc[4] = {};

    for (int kt = 0; kt < FIN; kt += 32) {
        // ---- A fragment: 16x32 bf16 from LDS (ds_load_b128 x2) ----
        v8bf alo = *(const v8bf*)(lx + lrow + kt + aoff);        // K = kt+aoff .. +7
        v8bf ahi = *(const v8bf*)(lx + lrow + kt + aoff + 8);    // K = kt+aoff+8 .. +15
        v16bf a;
        #pragma unroll
        for (int i = 0; i < 8; ++i) { a[i] = alo[i]; a[i + 8] = ahi[i]; }

        // ---- B fragments: 32x16 bf16; B[k,n] = W[n,k]; lanes<16 K=kt..+15, lanes>=16 K=kt+16..+31
        const int boff = kt + (hi << 4);
        #pragma unroll
        for (int t = 0; t < 4; ++t) {
            const int col = col0 + (t << 4) + (lane & 15);       // weight row
            __builtin_prefetch(W + (size_t)col * FIN + boff + 32, 0, 3);  // global_prefetch_b8
            v16bf b = *(const v16bf*)(W + (size_t)col * FIN + boff);
            acc[t] = __builtin_amdgcn_wmma_f32_16x16x32_bf16(
                         false, a, false, b, (short)0, acc[t], false, false);
        }
    }

    // ---- store: C/D layout, VGPR r -> M = r + 8*hi, N = lane&15 ----
    const int mbase = (mt << 4) + (hi << 3);
    #pragma unroll
    for (int t = 0; t < 4; ++t) {
        const int colg = col0 + (t << 4) + (lane & 15);
        float* dst = (colg < HFO) ? proj : skip;
        const int c = (colg < HFO) ? colg : colg - HFO;
        #pragma unroll
        for (int r = 0; r < 8; ++r) {
            const int m = mbase + r;
            if (m < nrows) dst[(size_t)m * HFO + c] = acc[t][r];
        }
    }
}

// ---------------- K2: attention scores per (node, head) ----------------
__global__ void scores_kernel(const float* __restrict__ proj,
                              const float* __restrict__ a_src, const float* __restrict__ a_trg,
                              float* __restrict__ ssrc, float* __restrict__ strg, int n) {
    int tid = blockIdx.x * blockDim.x + threadIdx.x;
    if (tid >= n * HH) return;
    const int node = tid >> 3, h = tid & 7;
    const float* p  = proj  + (size_t)node * HFO + h * FO;
    const float* as = a_src + h * FO;
    const float* at = a_trg + h * FO;
    float s1 = 0.f, s2 = 0.f;
    #pragma unroll
    for (int f = 0; f < FO; ++f) { float v = p[f]; s1 += v * as[f]; s2 += v * at[f]; }
    ssrc[tid] = s1;
    strg[tid] = s2;
}

// ---------------- K3: global max of leaky_relu(edge scores) ----------------
__global__ __launch_bounds__(256) void edge_max_kernel(
        const long long* __restrict__ ei, const float* __restrict__ ssrc,
        const float* __restrict__ strg, unsigned* __restrict__ gmax, int ne) {
    const int e = blockIdx.x * blockDim.x + threadIdx.x;
    float m = -3.0e38f;
    if (e < ne) {
        const int s = (int)ei[e], t = (int)ei[ne + e];
        #pragma unroll
        for (int h = 0; h < HH; ++h) {
            float v = ssrc[s * HH + h] + strg[t * HH + h];
            v = (v > 0.f) ? v : 0.2f * v;                    // leaky_relu
            m = fmaxf(m, v);
        }
    }
    __shared__ float red[256];
    red[threadIdx.x] = m;
    __syncthreads();
    for (int st = 128; st > 0; st >>= 1) {
        if (threadIdx.x < st) red[threadIdx.x] = fmaxf(red[threadIdx.x], red[threadIdx.x + st]);
        __syncthreads();
    }
    if (threadIdx.x == 0) atomicMax(gmax, fmap(red[0]));
}

// ---------------- K4: exp + denominator segment-sum ----------------
__global__ __launch_bounds__(256) void edge_exp_kernel(
        const long long* __restrict__ ei, const float* __restrict__ ssrc,
        const float* __restrict__ strg, const unsigned* __restrict__ gmax,
        float* __restrict__ exps, float* __restrict__ denom, int ne) {
    const int e = blockIdx.x * blockDim.x + threadIdx.x;
    if (e >= ne) return;
    const float gm = funmap(*gmax);
    const int s = (int)ei[e], t = (int)ei[ne + e];
    #pragma unroll
    for (int h = 0; h < HH; ++h) {
        float v = ssrc[s * HH + h] + strg[t * HH + h];
        v = (v > 0.f) ? v : 0.2f * v;
        const float ex = __expf(v - gm);
        exps[(size_t)e * HH + h] = ex;
        atomic_fadd(&denom[t * HH + h], ex);
    }
}

// ---------------- K5: weighted scatter-aggregation (wave per edge, lane per FO) ----------------
__global__ __launch_bounds__(256) void edge_agg_kernel(
        const long long* __restrict__ ei, const float* __restrict__ proj,
        const float* __restrict__ exps, const float* __restrict__ denom,
        float* __restrict__ out, int ne) {
    const int e = blockIdx.x * 8 + (threadIdx.x >> 5);
    const int f = threadIdx.x & 31;
    if (e >= ne) return;
    const int s = (int)ei[e], t = (int)ei[ne + e];
    const float* ps = proj + (size_t)s * HFO;
    float*       po = out  + (size_t)t * HFO;
    #pragma unroll
    for (int h = 0; h < HH; ++h) {
        const float w = exps[(size_t)e * HH + h] / (denom[t * HH + h] + 1e-16f);
        atomic_fadd(&po[h * FO + f], ps[h * FO + f] * w);
    }
}

// ---------------- K6: bias + ELU finalize ----------------
__global__ void finalize_kernel(float* __restrict__ out, const float* __restrict__ bias, long n) {
    long i = (long)blockIdx.x * blockDim.x + threadIdx.x;
    if (i >= n) return;
    float v = out[i] + bias[i & (HFO - 1)];
    out[i] = (v > 0.f) ? v : (__expf(v) - 1.0f);
}

// ---------------- launch ----------------
extern "C" void kernel_launch(void* const* d_in, const int* in_sizes, int n_in,
                              void* d_out, int out_size, void* d_ws, size_t ws_size,
                              hipStream_t stream) {
    const float*     x     = (const float*)d_in[0];
    const long long* ei    = (const long long*)d_in[1];
    const float*     Wp    = (const float*)d_in[2];
    const float*     a_src = (const float*)d_in[3];
    const float*     a_trg = (const float*)d_in[4];
    const float*     Wskip = (const float*)d_in[5];
    const float*     bias  = (const float*)d_in[6];
    float* out = (float*)d_out;

    const int nn = in_sizes[0] / FIN;     // 50000
    const int ne = in_sizes[1] / 2;       // 800000

    // workspace layout (256B aligned)
    char* ws = (char*)d_ws;
    size_t off = 0;
    auto take = [&](size_t bytes) { void* p = ws + off; off = (off + bytes + 255) & ~(size_t)255; return p; };
    __bf16*   xb    = (__bf16*)take((size_t)nn * FIN * sizeof(__bf16));
    __bf16*   Wcat  = (__bf16*)take((size_t)NCOL * FIN * sizeof(__bf16));
    float*    proj  = (float*) take((size_t)nn * HFO * sizeof(float));
    float*    ssrc  = (float*) take((size_t)nn * HH * sizeof(float));
    float*    strg  = (float*) take((size_t)nn * HH * sizeof(float));
    float*    exps  = (float*) take((size_t)ne * HH * sizeof(float));
    float*    denom = (float*) take((size_t)nn * HH * sizeof(float));
    unsigned* gmax  = (unsigned*)take(sizeof(unsigned));

    // zero-init accumulators (memset nodes are graph-capturable)
    hipMemsetAsync(denom, 0, (size_t)nn * HH * sizeof(float), stream);
    hipMemsetAsync(gmax, 0, sizeof(unsigned), stream);   // 0 < fmap(any finite float incl. negatives)

    // K0: converts
    cvt_bf16_kernel<<<2048, 256, 0, stream>>>(Wp,    Wcat,                 (long)HFO * FIN);
    cvt_bf16_kernel<<<2048, 256, 0, stream>>>(Wskip, Wcat + (size_t)HFO * FIN, (long)HFO * FIN);
    cvt_bf16_kernel<<<4096, 256, 0, stream>>>(x,     xb,                   (long)nn * FIN);

    // K1: fused WMMA GEMM (proj -> ws, skip -> d_out)
    const int mtiles = (nn + 15) / 16;   // 3125
    gemm_wmma_kernel<<<mtiles, 256, 0, stream>>>(xb, Wcat, proj, out, nn);

    // K2: scores
    scores_kernel<<<(nn * HH + 255) / 256, 256, 0, stream>>>(proj, a_src, a_trg, ssrc, strg, nn);

    // K3: global max
    edge_max_kernel<<<(ne + 255) / 256, 256, 0, stream>>>(ei, ssrc, strg, gmax, ne);

    // K4: exp + denom
    edge_exp_kernel<<<(ne + 255) / 256, 256, 0, stream>>>(ei, ssrc, strg, gmax, exps, denom, ne);

    // K5: aggregation into d_out (already holds skip)
    edge_agg_kernel<<<(ne + 7) / 8, 256, 0, stream>>>(ei, proj, exps, denom, out, ne);

    // K6: bias + ELU
    finalize_kernel<<<((long)nn * HFO + 255) / 256, 256, 0, stream>>>(out, bias, (long)nn * HFO);
}
